// VanRossumLoss_23940147708352
// MI455X (gfx1250) — matrix operand order repaired and checked by reading, not compiled, not run on hardware
//
#include <hip/hip_runtime.h>

typedef __attribute__((ext_vector_type(16))) _Float16 v16h;
typedef __attribute__((ext_vector_type(8)))  _Float16 v8h;
typedef __attribute__((ext_vector_type(8)))  float    v8f;

#define TT    2000   // input time steps
#define NN    512    // neurons
#define BB    32     // batch
#define TOUT  2001   // conv output length (same-pad, even K)
#define SPAN  512    // output columns per LDS window
#define NSPAN 4      // 4*512 = 2048 >= 2001
#define WINDOW 640   // staged input samples per window (512 + 114 halo, rounded)
#define LROW  648    // padded LDS row length in halves (1296 B, 16B aligned, bank-spread)

__global__ __launch_bounds__(256) void vanrossum_wmma_kernel(
    const float* __restrict__ out_sp,
    const float* __restrict__ tgt_sp,
    const float* __restrict__ prev_out,
    const float* __restrict__ prev_tgt,
    float* __restrict__ loss)
{
  __shared__ __align__(16) _Float16 sdiff[16 * LROW];   // [n_local][time_in_window]
  __shared__ __align__(16) _Float16 sB[4 * 32 * 16];    // [chunk][lane][elem] B fragments
  __shared__ float sred[8];

  const int tid  = threadIdx.x;
  const int lane = tid & 31;
  const int wid  = tid >> 5;
  const int seq_base = blockIdx.x * 16;     // 1024 blocks x 16 seqs = 16384
  const int b      = seq_base >> 9;         // /512 (16 | 512 so one batch per block)
  const int n_base = seq_base & 511;

  const float rstep = __expf(-0.05f);                          // exp(-dt/tau)
  const float knorm = (1.0f - rstep) / (1.0f - __expf(-5.0f)); // 1/sum(exp(-t/tau))

  const int ncol  = lane & 15;              // B col / D col (time within tile)
  const int mrow  = lane & 15;              // A row (seq within tile)
  const int khalf = (lane >= 16) ? 16 : 0;

  // ---- B fragments: banded kernel matrix, staged ONCE into LDS so the
  // compute loop reloads cheap ds_load_b128s instead of rematerializing
  // exp/cndmask chains between WMMAs (seen in round-1 disasm). ----
  if (wid == 0) {
    #pragma unroll
    for (int c = 0; c < 4; ++c) {
      #pragma unroll
      for (int e = 0; e < 16; ++e) {
        int kk = c * 32 + khalf + e;   // global K index in the 128-wide window
        int j  = kk - ncol;            // kernel tap index
        float v = (j >= 0 && j < 100) ? knorm * __expf(-0.05f * (float)j) : 0.0f;
        sB[(c * 32 + lane) * 16 + e] = (_Float16)v;
      }
    }
  }

  // ---- prev-state carry differences (C/D layout: row M = r + 8*(lane>=16)) ----
  const int mhalf = (lane >= 16) ? 8 : 0;
  float pdiff[8];
  #pragma unroll
  for (int r = 0; r < 8; ++r) {
    int s = seq_base + r + mhalf;
    pdiff[r] = prev_out[s] - prev_tgt[s];
  }

  float lsum = 0.0f;

  #pragma unroll 1
  for (int sp = 0; sp < NSPAN; ++sp) {
    const int tin0 = sp * SPAN - 50; // first staged input t (window halo = pad)

    // ---- cooperative, coalesced load of 16 x WINDOW f16 differences into LDS ----
    #pragma unroll 4
    for (int it = 0; it < (16 * WINDOW) / 256; ++it) {
      int idx = it * 256 + tid;
      int nl  = idx & 15;            // consecutive lanes -> consecutive neurons (64B rows)
      int i   = idx >> 4;
      int tg  = tin0 + i;
      float v = 0.0f;
      if (tg >= 0 && tg < TT) {
        size_t off = ((size_t)(b * TT + tg)) * NN + (size_t)(n_base + nl);
        v = out_sp[off] - tgt_sp[off];
      }
      sdiff[nl * LROW + i] = (_Float16)v;
    }
    __syncthreads();   // covers both sdiff and (first iteration) sB

    // ---- each wave computes 4 of the 32 16-column tiles of this window ----
    #pragma unroll 1
    for (int tt = 0; tt < 4; ++tt) {
      int tile = wid + tt * 8;                // 0..31
      int t0   = sp * SPAN + tile * 16;       // first global output column
      if (t0 < TOUT) {
        v8f acc = {};
        #pragma unroll
        for (int c = 0; c < 4; ++c) {
          // A 16-bit 16x32 layout (matches B): lane<16 elems {0..15}->K0..15;
          // lane>=16 elems -> K16..31. Two aligned 16B LDS reads each.
          int i0 = tile * 16 + c * 32 + ((lane >= 16) ? 16 : 0);
          const _Float16* rowp = &sdiff[mrow * LROW + i0];
          v8h alo = *(const v8h*)(rowp);
          v8h ahi = *(const v8h*)(rowp + 8);
          const _Float16* bp = &sB[(c * 32 + lane) * 16];
          v8h blo = *(const v8h*)(bp);
          v8h bhi = *(const v8h*)(bp + 8);
          v16h a, bf;
          #pragma unroll
          for (int e = 0; e < 8; ++e) {
            a[e]  = alo[e]; a[8 + e]  = ahi[e];
            bf[e] = blo[e]; bf[8 + e] = bhi[e];
          }
          acc = __builtin_amdgcn_wmma_f32_16x16x32_f16(
              false, a, false, bf, (short)0, acc, false, false);
        }
        int tg = t0 + ncol;                   // this lane's output column
        if (tg < TOUT) {
          float dec = __expf(-0.05f * (float)tg);
          #pragma unroll
          for (int r = 0; r < 8; ++r) {
            float ev = acc[r] + pdiff[r] * dec;
            lsum += ev * ev;
          }
        }
      }
    }
    __syncthreads();  // LDS window reused by next span
  }

  // ---- wave32 shuffle reduction, then one atomic per block ----
  #pragma unroll
  for (int off = 16; off > 0; off >>= 1)
    lsum += __shfl_xor(lsum, off, 32);
  if (lane == 0) sred[wid] = lsum;
  __syncthreads();
  if (tid == 0) {
    float s = 0.0f;
    #pragma unroll
    for (int w = 0; w < 8; ++w) s += sred[w];
    atomicAdd(loss, s * (1.0f / ((float)(BB * NN) * (float)TOUT)));
  }
}

extern "C" void kernel_launch(void* const* d_in, const int* in_sizes, int n_in,
                              void* d_out, int out_size, void* d_ws, size_t ws_size,
                              hipStream_t stream) {
  (void)in_sizes; (void)n_in; (void)d_ws; (void)ws_size;
  const float* out_sp   = (const float*)d_in[0];
  const float* tgt_sp   = (const float*)d_in[1];
  const float* prev_out = (const float*)d_in[2];
  const float* prev_tgt = (const float*)d_in[3];
  float* loss = (float*)d_out;

  hipMemsetAsync(loss, 0, (size_t)out_size * sizeof(float), stream); // capture-safe
  dim3 grid((BB * NN) / 16);   // 1024 blocks, one per 16-sequence group
  dim3 block(256);             // 8 wave32 waves
  vanrossum_wmma_kernel<<<grid, block, 0, stream>>>(out_sp, tgt_sp, prev_out, prev_tgt, loss);
}
